// DependencyEncoder3_8091718386295
// MI455X (gfx1250) — compile-verified, matrix operand order checked by reading, not run on hardware
//
#include <hip/hip_runtime.h>

// ---------------------------------------------------------------------------
// DependencyEncoder (TreeLSTM) for gfx1250 — WMMA f16-in/f32-acc GEMMs.
//   D=512, gates G=2048, K(branch)=8, P=8 dep types.
//   Levels: OFF = {0,1,9,73,585,4681,37449}
// ---------------------------------------------------------------------------

typedef __attribute__((ext_vector_type(16))) _Float16 v16h;
typedef __attribute__((ext_vector_type(8)))  _Float16 v8h;
typedef __attribute__((ext_vector_type(8)))  float    v8f;
typedef __attribute__((ext_vector_type(4)))  float    v4f;

#define DD   512
#define GG   2048
#define KB   8
#define PP   8
#define MAXC 32768

__device__ __forceinline__ v16h cat16(v8h a, v8h b) {
  return __builtin_shufflevector(a, b, 0,1,2,3,4,5,6,7,8,9,10,11,12,13,14,15);
}

// f16 fragment: lane's two 8-half chunks at +0 and +16 halves (ISA 16-bit layout).
__device__ __forceinline__ v16h ld16(const _Float16* p) {
  return cat16(*(const v8h*)p, *(const v8h*)(p + 16));
}

// f32 source fragment, converted to f16 on the fly (same chunk layout).
__device__ __forceinline__ v16h ld32(const float* p) {
  v4f a0 = *(const v4f*)(p),      a1 = *(const v4f*)(p + 4);
  v4f b0 = *(const v4f*)(p + 16), b1 = *(const v4f*)(p + 20);
  v16h r;
#pragma unroll
  for (int i = 0; i < 4; ++i) {
    r[i]      = (_Float16)a0[i];
    r[i + 4]  = (_Float16)a1[i];
    r[i + 8]  = (_Float16)b0[i];
    r[i + 12] = (_Float16)b1[i];
  }
  return r;
}

__device__ __forceinline__ v8f wmma16(v16h a, v16h b, v8f c) {
  return __builtin_amdgcn_wmma_f32_16x16x32_f16(false, a, false, b, (short)0, c,
                                                false, false);
}

__device__ __forceinline__ float sigmoidf_(float x) {
  return 1.0f / (1.0f + __expf(-x));
}
__device__ __forceinline__ float tanhf_(float x) {
  // 1 - 2/(e^{2x}+1): saturates cleanly at +/-1 for large |x|
  return 1.0f - 2.0f / (__expf(2.0f * x) + 1.0f);
}

// ---------------------------------------------------------------------------
// Kernel 1: f32 -> f16 weight conversion + combined bias (b_ih + b_hh).
// ---------------------------------------------------------------------------
__global__ void k_convert(const float* __restrict__ Wih, const float* __restrict__ Whh,
                          const float* __restrict__ Wdep,
                          const float* __restrict__ bih, const float* __restrict__ bhh,
                          _Float16* __restrict__ Wih16, _Float16* __restrict__ Whh16,
                          _Float16* __restrict__ Wdep16, float* __restrict__ bias) {
  const int n1 = GG * DD;            // W_ih
  const int n2 = n1 + GG * DD;       // W_hh
  const int n3 = n2 + PP * DD * DD;  // W_dep
  const int total = n3 + GG;         // bias
  for (int i = blockIdx.x * blockDim.x + threadIdx.x; i < total;
       i += gridDim.x * blockDim.x) {
    if (i < n1)       Wih16[i]       = (_Float16)Wih[i];
    else if (i < n2)  Whh16[i - n1]  = (_Float16)Whh[i - n1];
    else if (i < n3)  Wdep16[i - n2] = (_Float16)Wdep[i - n2];
    else              bias[i - n3]   = bih[i - n3] + bhh[i - n3];
  }
}

// ---------------------------------------------------------------------------
// Kernel 2: leaves. h=c=0, seqlen=1 -> z = sigmoid(i) * tanh(g), gates from
// x @ W_ih.T + b only (gate i = cols 0..511, gate g = cols 1024..1535).
// Block = 16 leaves x 512 dims, 8 waves each own 64 dims (4 col-tiles).
// ---------------------------------------------------------------------------
__global__ __launch_bounds__(256) void k_leaf(const float* __restrict__ embLeaf,
                                              const _Float16* __restrict__ Wih16,
                                              const float* __restrict__ bias,
                                              _Float16* __restrict__ zout) {
  const int tid = threadIdx.x, wave = tid >> 5, lane = tid & 31;
  const int laneHi = lane >> 4, l15 = lane & 15, kOff = laneHi * 8;
  const int tile = blockIdx.x * 16;

  const float* arow = embLeaf + (size_t)(tile + l15) * DD + kOff;

  v8f ai[4] = {}, ag[4] = {};
  const _Float16* bi[4];
  const _Float16* bg[4];
  int colN[4];
#pragma unroll
  for (int nt = 0; nt < 4; ++nt) {
    colN[nt] = wave * 64 + nt * 16 + l15;
    bi[nt] = Wih16 + ((size_t)(colN[nt])) * DD + kOff;          // gate i
    bg[nt] = Wih16 + ((size_t)(1024 + colN[nt])) * DD + kOff;   // gate g
  }
  for (int kk = 0; kk < DD; kk += 32) {
    v16h a = ld32(arow + kk);
#pragma unroll
    for (int nt = 0; nt < 4; ++nt) {
      ai[nt] = wmma16(a, ld16(bi[nt] + kk), ai[nt]);
      ag[nt] = wmma16(a, ld16(bg[nt] + kk), ag[nt]);
    }
  }
#pragma unroll
  for (int nt = 0; nt < 4; ++nt) {
    float bvi = bias[colN[nt]], bvg = bias[1024 + colN[nt]];
#pragma unroll
    for (int j = 0; j < 8; ++j) {
      float cv = sigmoidf_(ai[nt][j] + bvi) * tanhf_(ag[nt][j] + bvg);
      int r = tile + j + laneHi * 8;
      zout[(size_t)r * DD + colN[nt]] = (_Float16)cv;
    }
  }
}

// ---------------------------------------------------------------------------
// Dep-type bucketing (grouped-GEMM prep).
// ---------------------------------------------------------------------------
__global__ void k_zero_counts(int* counts) {
  if (threadIdx.x < PP) counts[threadIdx.x] = 0;
}

__global__ void k_scatter(const int* __restrict__ dep, int n,
                          int* __restrict__ buckets, int* __restrict__ counts) {
  int i = blockIdx.x * blockDim.x + threadIdx.x;
  if (i < n) {
    int p = dep[i];
    int slot = atomicAdd(&counts[p], 1);
    buckets[p * MAXC + slot] = i;
  }
}

// ---------------------------------------------------------------------------
// Kernel 3: grouped dep transform  zc[c] = z[c] @ W_dep[p_c].T + b_dep[p_c].
// grid = (ceil(nch/16), P); block handles one 16-row tile of bucket p.
// Rows gathered/scattered through bucket indices.
// ---------------------------------------------------------------------------
__global__ __launch_bounds__(256) void k_dep(const _Float16* __restrict__ zin,
                                             const _Float16* __restrict__ Wdep16,
                                             const float* __restrict__ bdep,
                                             const int* __restrict__ buckets,
                                             const int* __restrict__ counts,
                                             _Float16* __restrict__ zc) {
  const int p = blockIdx.y;
  const int cnt = counts[p];
  const int tile = blockIdx.x * 16;
  if (tile >= cnt) return;  // uniform per block

  const int* bkt = buckets + p * MAXC;
  const int tid = threadIdx.x, wave = tid >> 5, lane = tid & 31;
  const int laneHi = lane >> 4, l15 = lane & 15, kOff = laneHi * 8;

  int rA = tile + l15;
  if (rA >= cnt) rA = cnt - 1;  // pad with duplicate row (store predicated)
  const _Float16* arow = zin + (size_t)bkt[rA] * DD + kOff;
  const _Float16* wbase = Wdep16 + (size_t)p * DD * DD;

  v8f acc[4] = {};
  const _Float16* brow[4];
  int colN[4];
#pragma unroll
  for (int nt = 0; nt < 4; ++nt) {
    colN[nt] = wave * 64 + nt * 16 + l15;
    brow[nt] = wbase + (size_t)colN[nt] * DD + kOff;
  }
  for (int kk = 0; kk < DD; kk += 32) {
    v16h a = ld16(arow + kk);
#pragma unroll
    for (int nt = 0; nt < 4; ++nt)
      acc[nt] = wmma16(a, ld16(brow[nt] + kk), acc[nt]);
  }
#pragma unroll
  for (int nt = 0; nt < 4; ++nt) {
    float bv = bdep[p * DD + colN[nt]];
#pragma unroll
    for (int j = 0; j < 8; ++j) {
      int r = tile + j + laneHi * 8;
      if (r < cnt) {
        int child = bkt[r];
        zc[(size_t)child * DD + colN[nt]] = (_Float16)(acc[nt][j] + bv);
      }
    }
  }
}

// ---------------------------------------------------------------------------
// Kernel 4: 9-step LSTM over [zc(8 children), x], final c out.
// Block owns 16 nodes; h double-buffered in LDS (f16); c lives in the WMMA
// accumulator layout in registers. Wave w owns cell dims [w*64, w*64+64),
// i.e. for each 16-dim group the 4 gate column-tiles d, d+512, d+1024, d+1536.
// ---------------------------------------------------------------------------
__global__ __launch_bounds__(256) void k_lstm(const _Float16* __restrict__ zc,
                                              const float* __restrict__ emb, int embOff,
                                              int m,
                                              const _Float16* __restrict__ Wih16,
                                              const _Float16* __restrict__ Whh16,
                                              const float* __restrict__ bias,
                                              _Float16* __restrict__ zout,
                                              float* __restrict__ outF) {
  __shared__ __align__(16) _Float16 hbuf[2][16 * DD];

  const int tid = threadIdx.x, wave = tid >> 5, lane = tid & 31;
  const int laneHi = lane >> 4, l15 = lane & 15, kOff = laneHi * 8;
  const int tile = blockIdx.x * 16;

  for (int i = tid; i < 16 * DD; i += 256) hbuf[0][i] = (_Float16)0.0f;
  __syncthreads();

  int rowNode = tile + l15;
  if (rowNode >= m) rowNode = m - 1;  // padded rows (stores predicated)

  v8f c[4];
#pragma unroll
  for (int cg = 0; cg < 4; ++cg) c[cg] = v8f{};

  for (int t = 0; t < 9; ++t) {
    const _Float16* arow16 = (t < 8)
        ? zc + ((size_t)rowNode * KB + t) * DD + kOff : nullptr;
    const float* arow32 = (t == 8)
        ? emb + (size_t)(embOff + rowNode) * DD + kOff : nullptr;
    const _Float16* hrow = &hbuf[t & 1][l15 * DD + kOff];

    for (int cg = 0; cg < 4; ++cg) {
      const int colN = (wave * 4 + cg) * 16 + l15;
      v8f acc[4] = {};
      const _Float16* bW[4];
      const _Float16* bU[4];
#pragma unroll
      for (int q = 0; q < 4; ++q) {
        bW[q] = Wih16 + (size_t)(q * DD + colN) * DD + kOff;
        bU[q] = Whh16 + (size_t)(q * DD + colN) * DD + kOff;
      }
      // gates += z_t @ W_ih.T
      for (int kk = 0; kk < DD; kk += 32) {
        v16h a = (t < 8) ? ld16(arow16 + kk) : ld32(arow32 + kk);
#pragma unroll
        for (int q = 0; q < 4; ++q)
          acc[q] = wmma16(a, ld16(bW[q] + kk), acc[q]);
      }
      // gates += h @ W_hh.T  (h==0 at t=0; LDS pre-zeroed)
      for (int kk = 0; kk < DD; kk += 32) {
        v16h a = ld16(hrow + kk);
#pragma unroll
        for (int q = 0; q < 4; ++q)
          acc[q] = wmma16(a, ld16(bU[q] + kk), acc[q]);
      }
      const float bi = bias[colN],       bf = bias[DD + colN];
      const float bg = bias[2 * DD + colN], bo = bias[3 * DD + colN];
      v8f cc = c[cg];
#pragma unroll
      for (int j = 0; j < 8; ++j) {
        float iv = sigmoidf_(acc[0][j] + bi);
        float fv = sigmoidf_(acc[1][j] + bf);
        float gv = tanhf_(acc[2][j] + bg);
        float ov = sigmoidf_(acc[3][j] + bo);
        float cn = fv * cc[j] + iv * gv;
        cc[j] = cn;
        float hn = ov * tanhf_(cn);
        int r = j + laneHi * 8;
        hbuf[(t + 1) & 1][r * DD + colN] = (_Float16)hn;
      }
      c[cg] = cc;
    }
    __syncthreads();
  }

  // z = final c
#pragma unroll
  for (int cg = 0; cg < 4; ++cg) {
    const int colN = (wave * 4 + cg) * 16 + l15;
#pragma unroll
    for (int j = 0; j < 8; ++j) {
      int r = tile + j + laneHi * 8;
      if (r < m) {
        zout[(size_t)r * DD + colN] = (_Float16)c[cg][j];
        if (outF) outF[(size_t)r * DD + colN] = c[cg][j];
      }
    }
  }
}

// ---------------------------------------------------------------------------
// Host launcher
// ---------------------------------------------------------------------------
extern "C" void kernel_launch(void* const* d_in, const int* in_sizes, int n_in,
                              void* d_out, int out_size, void* d_ws, size_t ws_size,
                              hipStream_t stream) {
  const float* emb  = (const float*)d_in[0];
  const int*   dep  = (const int*)d_in[1];
  const float* Wdep = (const float*)d_in[2];
  const float* bdep = (const float*)d_in[3];
  const float* Wih  = (const float*)d_in[4];
  const float* Whh  = (const float*)d_in[5];
  const float* bih  = (const float*)d_in[6];
  const float* bhh  = (const float*)d_in[7];
  float* out = (float*)d_out;

  char* ws = (char*)d_ws;
  size_t off = 0;
  auto take = [&](size_t bytes) -> char* {
    char* p = ws + off;
    off = (off + bytes + 255) & ~(size_t)255;
    return p;
  };
  _Float16* Wih16  = (_Float16*)take((size_t)GG * DD * 2);
  _Float16* Whh16  = (_Float16*)take((size_t)GG * DD * 2);
  _Float16* Wdep16 = (_Float16*)take((size_t)PP * DD * DD * 2);
  float*    bias   = (float*)take((size_t)GG * 4);
  int*      counts = (int*)take(PP * 4);
  int*      bucket = (int*)take((size_t)PP * MAXC * 4);
  _Float16* zc     = (_Float16*)take((size_t)MAXC * DD * 2);
  _Float16* zb0    = (_Float16*)take((size_t)MAXC * DD * 2);
  _Float16* zb1    = (_Float16*)take((size_t)4096 * DD * 2);

  static const int OFFh[7] = {0, 1, 9, 73, 585, 4681, 37449};

  k_convert<<<2048, 256, 0, stream>>>(Wih, Whh, Wdep, bih, bhh,
                                      Wih16, Whh16, Wdep16, bias);

  // Leaves: 32768 nodes, z -> zb0 (f16)
  k_leaf<<<32768 / 16, 256, 0, stream>>>(emb + (size_t)OFFh[5] * DD,
                                         Wih16, bias, zb0);

  _Float16* zin = zb0;
  _Float16* zo  = zb1;
  for (int l = 4; l >= 0; --l) {
    const int m   = OFFh[l + 1] - OFFh[l];
    const int nch = OFFh[l + 2] - OFFh[l + 1];

    k_zero_counts<<<1, 32, 0, stream>>>(counts);
    k_scatter<<<(nch + 255) / 256, 256, 0, stream>>>(dep + OFFh[l + 1], nch,
                                                     bucket, counts);
    dim3 dg((nch + 15) / 16, PP);
    k_dep<<<dg, 256, 0, stream>>>(zin, Wdep16, bdep, bucket, counts, zc);

    k_lstm<<<(m + 15) / 16, 256, 0, stream>>>(zc, emb, OFFh[l], m, Wih16, Whh16,
                                              bias, zo, (l == 0) ? out : nullptr);
    _Float16* t = zin; zin = zo; zo = t;
  }
}
